// MiMoV2DecoderLayer_37881611550761
// MI455X (gfx1250) — compile-verified, matrix operand order checked by reading, not run on hardware
//
#include <hip/hip_runtime.h>

// ---------------------------------------------------------------------------
// MiMoV2 decoder layer for gfx1250 (MI455X), bf16 WMMA path.
// Ping-pong double-buffered WMMA GEMMs (no register-rotation copies).
// ---------------------------------------------------------------------------

typedef __attribute__((ext_vector_type(16))) __bf16 v16bf;
typedef __attribute__((ext_vector_type(8)))  __bf16 v8bf;
typedef __attribute__((ext_vector_type(8)))  float  v8f;

static constexpr int Tt   = 1024;   // tokens (B*S)
static constexpr int Hh   = 2048;   // hidden
static constexpr int NH   = 16;
static constexpr int NKV  = 4;
static constexpr int HD   = 128;
static constexpr int Ee   = 16;     // experts
static constexpr int TOPK = 4;
static constexpr int DFF  = 1024;
static constexpr int Gg   = 4;      // groups
static constexpr int QKV_N = (NH + 2 * NKV) * HD;   // 3072

// ---------------- helpers ----------------

__device__ __forceinline__ unsigned short f2bf(float f) {
  unsigned int u = __float_as_uint(f);
  unsigned int r = u + 0x7FFFu + ((u >> 16) & 1u);   // RNE
  return (unsigned short)(r >> 16);
}
__device__ __forceinline__ float bf2f(unsigned short h) {
  return __uint_as_float(((unsigned int)h) << 16);
}

// lane holds 16 bf16 values: k = [8h, 8h+8) and [16+8h, 16+8h+8) relative to p
// (p already includes the +8h term). Two 16-byte loads.
__device__ __forceinline__ v16bf load_frag(const __bf16* p) {
  v8bf lo = *(const v8bf*)p;
  v8bf hi = *(const v8bf*)(p + 16);
  v16bf r;
#pragma unroll
  for (int i = 0; i < 8; ++i) { r[i] = lo[i]; r[i + 8] = hi[i]; }
  return r;
}

__device__ __forceinline__ v8f wmma_bf16(v16bf a, v16bf b, v8f c) {
  return __builtin_amdgcn_wmma_f32_16x16x32_bf16(false, a, false, b,
                                                 (short)0, c, false, false);
}

// ---------------- weight convert + transpose: [b][k][n] f32 -> [b][n][k] bf16

__global__ __launch_bounds__(256)
void convert_transpose_kernel(const float* __restrict__ src,
                              unsigned short* __restrict__ dst,
                              int Kd, int N, long total) {
  long stride = (long)gridDim.x * 256;
  for (long idx = (long)blockIdx.x * 256 + threadIdx.x; idx < total; idx += stride) {
    long bk = idx / N;
    int  n  = (int)(idx % N);
    long b  = bk / Kd;
    int  k  = (int)(bk % Kd);
    dst[(b * N + n) * (long)Kd + k] = f2bf(src[idx]);
  }
}

// ---------------- RMSNorm -> bf16 ----------------

__global__ __launch_bounds__(256)
void rmsnorm_kernel(const float* __restrict__ x, const float* __restrict__ w,
                    unsigned short* __restrict__ out) {
  __shared__ float red[8];
  int t = blockIdx.x;
  const float* xi = x + (long)t * Hh;
  float s = 0.f;
  for (int i = threadIdx.x; i < Hh; i += 256) { float v = xi[i]; s += v * v; }
#pragma unroll
  for (int m = 16; m >= 1; m >>= 1) s += __shfl_xor(s, m, 32);
  if ((threadIdx.x & 31) == 0) red[threadIdx.x >> 5] = s;
  __syncthreads();
  float tot = 0.f;
#pragma unroll
  for (int i = 0; i < 8; ++i) tot += red[i];
  float inv = rsqrtf(tot / (float)Hh + 1e-6f);
  for (int i = threadIdx.x; i < Hh; i += 256)
    out[(long)t * Hh + i] = f2bf(xi[i] * inv * w[i]);
}

// ---------------- generic bf16 WMMA GEMM -------------------------------------
// C[M,N] = A[M,K](bf16,row-major) * Bt[N,K](bf16) (+residual) -> f32 (+copy)
// one wave computes a 16x64 strip; K unrolled by 64 with ping-pong fragment
// sets so loads land directly in WMMA operand registers (no rotation movs).
// Requires Kd % 64 == 0 and Kd >= 128 (holds: 2048/1024).

__global__ __launch_bounds__(256)
void gemm_bf16_kernel(const unsigned short* __restrict__ A,
                      const unsigned short* __restrict__ Bt,
                      float* __restrict__ C, const float* __restrict__ residual,
                      float* __restrict__ C2, int M, int N, int Kd) {
  int wave = (blockIdx.x * 256 + threadIdx.x) >> 5;
  int lane = threadIdx.x & 31;
  int ntiles = N >> 6;
  int mt = wave / ntiles, nt = wave % ntiles;
  if (mt * 16 >= M) return;
  int r = lane & 15, h = lane >> 4;
  int row0 = mt * 16, col0 = nt * 64;

  const __bf16* arow = (const __bf16*)A + (long)(row0 + r) * Kd + 8 * h;
  const __bf16* bcol[4];
#pragma unroll
  for (int j = 0; j < 4; ++j)
    bcol[j] = (const __bf16*)Bt + (long)(col0 + j * 16 + r) * Kd + 8 * h;

  v8f acc[4] = {};
  v16bf a0 = load_frag(arow);
  v16bf b0[4];
#pragma unroll
  for (int j = 0; j < 4; ++j) b0[j] = load_frag(bcol[j]);

  v16bf a1; v16bf b1[4];
  int k0 = 0;
  for (; k0 < Kd - 64; k0 += 64) {
    // stage set1 (k0+32)
    a1 = load_frag(arow + k0 + 32);
#pragma unroll
    for (int j = 0; j < 4; ++j) b1[j] = load_frag(bcol[j] + k0 + 32);
    __builtin_prefetch(arow + k0 + 160, 0, 1);
    // compute set0 (k0)
#pragma unroll
    for (int j = 0; j < 4; ++j) acc[j] = wmma_bf16(a0, b0[j], acc[j]);
    // stage set0 (k0+64)
    a0 = load_frag(arow + k0 + 64);
#pragma unroll
    for (int j = 0; j < 4; ++j) b0[j] = load_frag(bcol[j] + k0 + 64);
    // compute set1 (k0+32)
#pragma unroll
    for (int j = 0; j < 4; ++j) acc[j] = wmma_bf16(a1, b1[j], acc[j]);
  }
  // tail: set0 holds Kd-64; stage and compute last two steps
  a1 = load_frag(arow + k0 + 32);
#pragma unroll
  for (int j = 0; j < 4; ++j) b1[j] = load_frag(bcol[j] + k0 + 32);
#pragma unroll
  for (int j = 0; j < 4; ++j) acc[j] = wmma_bf16(a0, b0[j], acc[j]);
#pragma unroll
  for (int j = 0; j < 4; ++j) acc[j] = wmma_bf16(a1, b1[j], acc[j]);

#pragma unroll
  for (int j = 0; j < 4; ++j) {
    int col = col0 + j * 16 + r;
#pragma unroll
    for (int rr = 0; rr < 8; ++rr) {
      int row = row0 + rr + 8 * h;
      float v = acc[j][rr];
      if (residual) v += residual[(long)row * N + col];
      C[(long)row * N + col] = v;
      if (C2) C2[(long)row * N + col] = v;
    }
  }
}

// ---------------- RoPE + pack Q/K/V into WMMA-friendly layouts ---------------
// Qb[NH][T][HD] bf16 (prescaled by HD^-0.5), Kb[NKV][T][HD], Vt[NKV][HD][T]

__global__ __launch_bounds__(256)
void rope_pack_kernel(const float* __restrict__ qkv, const int* __restrict__ pos,
                      unsigned short* __restrict__ Qb,
                      unsigned short* __restrict__ Kb,
                      unsigned short* __restrict__ Vt) {
  const long NQ = (long)Tt * NH * 64;
  const long NK = (long)Tt * NKV * 64;
  const long NV = (long)Tt * NKV * 128;
  long idx = (long)blockIdx.x * 256 + threadIdx.x;
  const float LOG_THETA = 13.815510558f;                 // ln(1e6)
  if (idx < NQ) {
    int d = (int)(idx & 63);
    long r2 = idx >> 6;
    int hh = (int)(r2 & 15);
    int t  = (int)(r2 >> 4);
    float p = (float)pos[t];
    float ang = p * __expf(-(float)d * (LOG_THETA / 64.f));
    float sn, cs; __sincosf(ang, &sn, &cs);
    float x1 = qkv[(long)t * QKV_N + hh * HD + d];
    float x2 = qkv[(long)t * QKV_N + hh * HD + d + 64];
    const float qs = 0.08838834764831845f;               // 128^-0.5
    Qb[((long)hh * Tt + t) * HD + d]      = f2bf((x1 * cs - x2 * sn) * qs);
    Qb[((long)hh * Tt + t) * HD + d + 64] = f2bf((x2 * cs + x1 * sn) * qs);
  } else if (idx < NQ + NK) {
    long i2 = idx - NQ;
    int d = (int)(i2 & 63);
    long r2 = i2 >> 6;
    int kv = (int)(r2 & 3);
    int t  = (int)(r2 >> 2);
    float p = (float)pos[t];
    float ang = p * __expf(-(float)d * (LOG_THETA / 64.f));
    float sn, cs; __sincosf(ang, &sn, &cs);
    float x1 = qkv[(long)t * QKV_N + NH * HD + kv * HD + d];
    float x2 = qkv[(long)t * QKV_N + NH * HD + kv * HD + d + 64];
    Kb[((long)kv * Tt + t) * HD + d]      = f2bf(x1 * cs - x2 * sn);
    Kb[((long)kv * Tt + t) * HD + d + 64] = f2bf(x2 * cs + x1 * sn);
  } else if (idx < NQ + NK + NV) {
    long i2 = idx - NQ - NK;
    int d = (int)(i2 & 127);
    long r2 = i2 >> 7;
    int kv = (int)(r2 & 3);
    int t  = (int)(r2 >> 2);
    Vt[((long)kv * HD + d) * Tt + t] =
        f2bf(qkv[(long)t * QKV_N + (NH + NKV) * HD + kv * HD + d]);
  }
}

// ---------------- flash attention, one wave per (head, 16-query tile) --------

__global__ __launch_bounds__(256)
void attn_kernel(const unsigned short* __restrict__ Qb,
                 const unsigned short* __restrict__ Kb,
                 const unsigned short* __restrict__ Vt,
                 unsigned short* __restrict__ attn_out) {
  __shared__ __align__(16) unsigned short pshm[8][16 * 32];
  int wave  = (blockIdx.x * 256 + threadIdx.x) >> 5;
  int wl    = (threadIdx.x >> 5);
  int lane  = threadIdx.x & 31;
  int head  = wave >> 6;
  int qt    = wave & 63;
  int kv    = head >> 2;                 // NH/NKV = 4
  int qbase = qt * 16;
  int r = lane & 15, h = lane >> 4;

  const __bf16* qrow = (const __bf16*)Qb + ((long)head * Tt + qbase + r) * HD + 8 * h;
  v16bf qf[4];
#pragma unroll
  for (int i = 0; i < 4; ++i) qf[i] = load_frag(qrow + 32 * i);

  const __bf16* kb_p = (const __bf16*)Kb + (long)kv * Tt * HD;
  const __bf16* vt_p = (const __bf16*)Vt + (long)kv * HD * Tt;

  v8f acc[8] = {};
  float m[8], l[8];
#pragma unroll
  for (int i = 0; i < 8; ++i) { m[i] = -1e30f; l[i] = 0.f; }

  int kend = qbase + 16;
  for (int kb = 0; kb < kend; kb += 32) {
    bool have2 = (kb + 16) < kend;
    v8f s0 = {}, s1 = {};
#pragma unroll
    for (int i = 0; i < 4; ++i) {
      v16bf b0 = load_frag(kb_p + (long)(kb + r) * HD + 32 * i + 8 * h);
      s0 = wmma_bf16(qf[i], b0, s0);
    }
    if (have2) {
#pragma unroll
      for (int i = 0; i < 4; ++i) {
        v16bf b1 = load_frag(kb_p + (long)(kb + 16 + r) * HD + 32 * i + 8 * h);
        s1 = wmma_bf16(qf[i], b1, s1);
      }
    }
    float mt_[8];
#pragma unroll
    for (int rr = 0; rr < 8; ++rr) {
      int q = qbase + rr + 8 * h;
      if (kb + r > q) s0[rr] = -1e30f;
      if (have2 && (kb + 16 + r) > q) s1[rr] = -1e30f;
      float v0 = s0[rr];
      if (have2) v0 = fmaxf(v0, s1[rr]);
      mt_[rr] = v0;
    }
#pragma unroll
    for (int msk = 1; msk < 16; msk <<= 1)
#pragma unroll
      for (int rr = 0; rr < 8; ++rr)
        mt_[rr] = fmaxf(mt_[rr], __shfl_xor(mt_[rr], msk, 32));
    float alpha[8], rs[8];
#pragma unroll
    for (int rr = 0; rr < 8; ++rr) {
      float nm = fmaxf(m[rr], mt_[rr]);
      alpha[rr] = __expf(m[rr] - nm);
      m[rr] = nm;
      s0[rr] = __expf(s0[rr] - nm);
      s1[rr] = have2 ? __expf(s1[rr] - nm) : 0.f;
      rs[rr] = s0[rr] + s1[rr];
    }
#pragma unroll
    for (int msk = 1; msk < 16; msk <<= 1)
#pragma unroll
      for (int rr = 0; rr < 8; ++rr)
        rs[rr] += __shfl_xor(rs[rr], msk, 32);
#pragma unroll
    for (int rr = 0; rr < 8; ++rr) l[rr] = l[rr] * alpha[rr] + rs[rr];
#pragma unroll
    for (int t8 = 0; t8 < 8; ++t8)
#pragma unroll
      for (int rr = 0; rr < 8; ++rr) acc[t8][rr] *= alpha[rr];

    // P (C-layout f32) -> bf16 A-layout via per-wave LDS tile [16 rows][32 keys]
    unsigned short* ps = &pshm[wl][0];
#pragma unroll
    for (int rr = 0; rr < 8; ++rr) {
      ps[(rr + 8 * h) * 32 + r]      = f2bf(s0[rr]);
      ps[(rr + 8 * h) * 32 + 16 + r] = f2bf(s1[rr]);
    }
    v16bf pf = load_frag((const __bf16*)(ps + r * 32 + 8 * h));
#pragma unroll
    for (int t8 = 0; t8 < 8; ++t8) {
      v16bf vf = load_frag(vt_p + (long)(t8 * 16 + r) * Tt + kb + 8 * h);
      acc[t8] = wmma_bf16(pf, vf, acc[t8]);
    }
  }
#pragma unroll
  for (int t8 = 0; t8 < 8; ++t8)
#pragma unroll
    for (int rr = 0; rr < 8; ++rr)
      attn_out[(long)(qbase + rr + 8 * h) * Hh + head * HD + t8 * 16 + r] =
          f2bf(acc[t8][rr] / l[rr]);
}

// ---------------- router: group-limited top-4, builds per-expert lists -------

__global__ void zero_counts_kernel(int* counts) {
  if (threadIdx.x < Ee) counts[threadIdx.x] = 0;
}

__global__ __launch_bounds__(256)
void router_kernel(const unsigned short* __restrict__ xn2,
                   const float* __restrict__ gate_w,
                   const float* __restrict__ gate_bias,
                   int* __restrict__ counts, int* __restrict__ etok,
                   float* __restrict__ ewt) {
  int t = blockIdx.x * 256 + threadIdx.x;
  if (t >= Tt) return;
  float sig[Ee], sb[Ee];
  for (int e = 0; e < Ee; ++e) {
    float s = 0.f;
    for (int d = 0; d < Hh; ++d)
      s += bf2f(xn2[(long)t * Hh + d]) * gate_w[(long)e * Hh + d];
    float sg = 1.f / (1.f + __expf(-s));
    sig[e] = sg;
    sb[e] = sg + gate_bias[e];
  }
  float gs[Gg];
#pragma unroll
  for (int g = 0; g < Gg; ++g) {
    float m1 = -1e30f, m2 = -1e30f;
#pragma unroll
    for (int j = 0; j < Ee / Gg; ++j) {
      float v = sb[g * (Ee / Gg) + j];
      if (v > m1) { m2 = m1; m1 = v; } else if (v > m2) { m2 = v; }
    }
    gs[g] = m1 + m2;
  }
  int g1 = 0;
  for (int g = 1; g < Gg; ++g) if (gs[g] > gs[g1]) g1 = g;
  float keep = gs[g1]; gs[g1] = -3e30f;
  int g2 = 0;
  for (int g = 1; g < Gg; ++g) if (gs[g] > gs[g2]) g2 = g;
  gs[g1] = keep;
  float sbm[Ee];
  for (int e = 0; e < Ee; ++e) {
    int g = e / (Ee / Gg);
    sbm[e] = (g == g1 || g == g2) ? sb[e] : -1e30f;
  }
  int isel[TOPK]; float wsel[TOPK]; float ssum = 0.f;
#pragma unroll
  for (int kk = 0; kk < TOPK; ++kk) {
    int bi = 0; float best = sbm[0];
    for (int e = 1; e < Ee; ++e) if (sbm[e] > best) { best = sbm[e]; bi = e; }
    isel[kk] = bi; wsel[kk] = sig[bi]; ssum += sig[bi]; sbm[bi] = -2e30f;
  }
#pragma unroll
  for (int kk = 0; kk < TOPK; ++kk) {
    float wv = wsel[kk] / (ssum + 1e-20f);
    int slot = atomicAdd(&counts[isel[kk]], 1);
    etok[isel[kk] * Tt + slot] = t;
    ewt[isel[kk] * Tt + slot] = wv;
  }
}

// ---------------- MoE grouped GEMMs ------------------------------------------
// gate_up: act[e][slot][f] = silu(g)*u; ping-pong K pipeline (H steps).

__global__ __launch_bounds__(256)
void moe_gateup_kernel(const unsigned short* __restrict__ xn2,
                       const unsigned short* __restrict__ guwt, // [E][2*DFF][H]
                       const int* __restrict__ counts, const int* __restrict__ etok,
                       unsigned short* __restrict__ act) {
  const int NTt = DFF / 64;   // 16
  const int MTt = Tt / 16;    // 64
  int wave = (blockIdx.x * 256 + threadIdx.x) >> 5;
  int lane = threadIdx.x & 31;
  int e = wave / (NTt * MTt);
  int rem = wave % (NTt * MTt);
  int mt = rem / NTt, nt = rem % NTt;
  int cnt = counts[e];
  if (mt * 16 >= cnt) return;
  int r = lane & 15, h = lane >> 4;
  int slot = mt * 16 + r;
  int sc = slot < cnt ? slot : cnt - 1;
  int tok = etok[e * Tt + sc];

  const __bf16* arow = (const __bf16*)xn2 + (long)tok * Hh + 8 * h;
  const __bf16* Bt = (const __bf16*)guwt + (long)e * 2 * DFF * Hh;
  const __bf16 *bg[4], *bu[4];
#pragma unroll
  for (int j = 0; j < 4; ++j) {
    bg[j] = Bt + (long)(nt * 64 + j * 16 + r) * Hh + 8 * h;
    bu[j] = Bt + (long)(DFF + nt * 64 + j * 16 + r) * Hh + 8 * h;
  }
  v8f ag[4] = {}, au[4] = {};
  v16bf a0 = load_frag(arow);
  v16bf g0[4], u0[4];
#pragma unroll
  for (int j = 0; j < 4; ++j) { g0[j] = load_frag(bg[j]); u0[j] = load_frag(bu[j]); }

  v16bf a1, g1v[4], u1v[4];
  int k0 = 0;
  for (; k0 < Hh - 64; k0 += 64) {
    a1 = load_frag(arow + k0 + 32);
#pragma unroll
    for (int j = 0; j < 4; ++j) {
      g1v[j] = load_frag(bg[j] + k0 + 32);
      u1v[j] = load_frag(bu[j] + k0 + 32);
    }
#pragma unroll
    for (int j = 0; j < 4; ++j) {
      ag[j] = wmma_bf16(a0, g0[j], ag[j]);
      au[j] = wmma_bf16(a0, u0[j], au[j]);
    }
    a0 = load_frag(arow + k0 + 64);
#pragma unroll
    for (int j = 0; j < 4; ++j) {
      g0[j] = load_frag(bg[j] + k0 + 64);
      u0[j] = load_frag(bu[j] + k0 + 64);
    }
#pragma unroll
    for (int j = 0; j < 4; ++j) {
      ag[j] = wmma_bf16(a1, g1v[j], ag[j]);
      au[j] = wmma_bf16(a1, u1v[j], au[j]);
    }
  }
  a1 = load_frag(arow + k0 + 32);
#pragma unroll
  for (int j = 0; j < 4; ++j) {
    g1v[j] = load_frag(bg[j] + k0 + 32);
    u1v[j] = load_frag(bu[j] + k0 + 32);
  }
#pragma unroll
  for (int j = 0; j < 4; ++j) {
    ag[j] = wmma_bf16(a0, g0[j], ag[j]);
    au[j] = wmma_bf16(a0, u0[j], au[j]);
  }
#pragma unroll
  for (int j = 0; j < 4; ++j) {
    ag[j] = wmma_bf16(a1, g1v[j], ag[j]);
    au[j] = wmma_bf16(a1, u1v[j], au[j]);
  }

#pragma unroll
  for (int j = 0; j < 4; ++j)
#pragma unroll
    for (int rr = 0; rr < 8; ++rr) {
      int sl = mt * 16 + rr + 8 * h;
      if (sl < cnt) {
        float g = ag[j][rr], u = au[j][rr];
        float v = g / (1.f + __expf(-g)) * u;      // silu(g)*u
        act[((long)e * Tt + sl) * DFF + nt * 64 + j * 16 + r] = f2bf(v);
      }
    }
}

// down: out[tok] += w * (act[e][slot] @ down_wt[e]),  down_wt [E][H][DFF]

__global__ __launch_bounds__(256)
void moe_down_kernel(const unsigned short* __restrict__ act,
                     const unsigned short* __restrict__ dwt,
                     const int* __restrict__ counts, const int* __restrict__ etok,
                     const float* __restrict__ ewt, float* __restrict__ out) {
  const int NTt = Hh / 64;    // 32
  const int MTt = Tt / 16;    // 64
  int wave = (blockIdx.x * 256 + threadIdx.x) >> 5;
  int lane = threadIdx.x & 31;
  int e = wave / (NTt * MTt);
  int rem = wave % (NTt * MTt);
  int mt = rem / NTt, nt = rem % NTt;
  int cnt = counts[e];
  if (mt * 16 >= cnt) return;
  int r = lane & 15, h = lane >> 4;

  const __bf16* arow = (const __bf16*)act + ((long)e * Tt + mt * 16 + r) * DFF + 8 * h;
  const __bf16* Bt = (const __bf16*)dwt + (long)e * Hh * DFF;
  const __bf16* bcol[4];
#pragma unroll
  for (int j = 0; j < 4; ++j)
    bcol[j] = Bt + (long)(nt * 64 + j * 16 + r) * DFF + 8 * h;

  v8f acc[4] = {};
  v16bf a0 = load_frag(arow);
  v16bf b0[4];
#pragma unroll
  for (int j = 0; j < 4; ++j) b0[j] = load_frag(bcol[j]);

  v16bf a1, b1[4];
  int k0 = 0;
  for (; k0 < DFF - 64; k0 += 64) {
    a1 = load_frag(arow + k0 + 32);
#pragma unroll
    for (int j = 0; j < 4; ++j) b1[j] = load_frag(bcol[j] + k0 + 32);
#pragma unroll
    for (int j = 0; j < 4; ++j) acc[j] = wmma_bf16(a0, b0[j], acc[j]);
    a0 = load_frag(arow + k0 + 64);
#pragma unroll
    for (int j = 0; j < 4; ++j) b0[j] = load_frag(bcol[j] + k0 + 64);
#pragma unroll
    for (int j = 0; j < 4; ++j) acc[j] = wmma_bf16(a1, b1[j], acc[j]);
  }
  a1 = load_frag(arow + k0 + 32);
#pragma unroll
  for (int j = 0; j < 4; ++j) b1[j] = load_frag(bcol[j] + k0 + 32);
#pragma unroll
  for (int j = 0; j < 4; ++j) acc[j] = wmma_bf16(a0, b0[j], acc[j]);
#pragma unroll
  for (int j = 0; j < 4; ++j) acc[j] = wmma_bf16(a1, b1[j], acc[j]);

#pragma unroll
  for (int j = 0; j < 4; ++j)
#pragma unroll
    for (int rr = 0; rr < 8; ++rr) {
      int sl = mt * 16 + rr + 8 * h;
      if (sl < cnt) {
        int tok = etok[e * Tt + sl];
        float w = ewt[e * Tt + sl];
        atomicAdd(&out[(long)tok * Hh + nt * 64 + j * 16 + r], w * acc[j][rr]);
      }
    }
}

// ---------------- launch ----------------

extern "C" void kernel_launch(void* const* d_in, const int* in_sizes, int n_in,
                              void* d_out, int out_size, void* d_ws, size_t ws_size,
                              hipStream_t stream) {
  (void)in_sizes; (void)n_in; (void)out_size; (void)ws_size;
  const float* hidden    = (const float*)d_in[0];
  const int*   positions = (const int*)d_in[1];
  const float* in_norm_w = (const float*)d_in[2];
  const float* post_norm = (const float*)d_in[3];
  const float* qkv_w     = (const float*)d_in[4];
  const float* o_w       = (const float*)d_in[5];
  const float* gate_w    = (const float*)d_in[6];
  const float* gate_bias = (const float*)d_in[7];
  const float* gate_up_w = (const float*)d_in[8];
  const float* down_w    = (const float*)d_in[9];
  float* out = (float*)d_out;

  char* wp = (char*)d_ws;
  auto alloc = [&](size_t bytes) -> void* {
    void* p = wp;
    wp += (bytes + 255) & ~(size_t)255;
    return p;
  };
  unsigned short* qkv_wt = (unsigned short*)alloc((size_t)QKV_N * Hh * 2);
  unsigned short* o_wt   = (unsigned short*)alloc((size_t)Hh * Hh * 2);
  unsigned short* gu_wt  = (unsigned short*)alloc((size_t)Ee * 2 * DFF * Hh * 2);
  unsigned short* dn_wt  = (unsigned short*)alloc((size_t)Ee * Hh * DFF * 2);
  unsigned short* xn1    = (unsigned short*)alloc((size_t)Tt * Hh * 2);
  float*          qkvo   = (float*)alloc((size_t)Tt * QKV_N * 4);
  unsigned short* Qb     = (unsigned short*)alloc((size_t)NH * Tt * HD * 2);
  unsigned short* Kb     = (unsigned short*)alloc((size_t)NKV * Tt * HD * 2);
  unsigned short* Vt     = (unsigned short*)alloc((size_t)NKV * HD * Tt * 2);
  unsigned short* attn   = (unsigned short*)alloc((size_t)Tt * Hh * 2);
  float*          hres   = (float*)alloc((size_t)Tt * Hh * 4);
  unsigned short* xn2    = (unsigned short*)alloc((size_t)Tt * Hh * 2);
  int*            counts = (int*)alloc(Ee * 4);
  int*            etok   = (int*)alloc((size_t)Ee * Tt * 4);
  float*          ewt    = (float*)alloc((size_t)Ee * Tt * 4);
  unsigned short* actb   = (unsigned short*)alloc((size_t)Ee * Tt * DFF * 2);

  // 1) weight convert+transpose (f32 [k][n] -> bf16 [n][k])
  convert_transpose_kernel<<<4096, 256, 0, stream>>>(qkv_w, qkv_wt, Hh, QKV_N,
                                                     (long)Hh * QKV_N);
  convert_transpose_kernel<<<4096, 256, 0, stream>>>(o_w, o_wt, Hh, Hh,
                                                     (long)Hh * Hh);
  convert_transpose_kernel<<<8192, 256, 0, stream>>>(gate_up_w, gu_wt, Hh, 2 * DFF,
                                                     (long)Ee * Hh * 2 * DFF);
  convert_transpose_kernel<<<8192, 256, 0, stream>>>(down_w, dn_wt, DFF, Hh,
                                                     (long)Ee * DFF * Hh);
  // 2) pre-attention norm
  rmsnorm_kernel<<<Tt, 256, 0, stream>>>(hidden, in_norm_w, xn1);
  // 3) QKV GEMM (1024x3072x2048)
  gemm_bf16_kernel<<<(Tt / 16) * (QKV_N / 64) / 8, 256, 0, stream>>>(
      xn1, qkv_wt, qkvo, nullptr, nullptr, Tt, QKV_N, Hh);
  // 4) RoPE + pack
  {
    long total = (long)Tt * NH * 64 + (long)Tt * NKV * 64 + (long)Tt * NKV * 128;
    rope_pack_kernel<<<(int)((total + 255) / 256), 256, 0, stream>>>(
        qkvo, positions, Qb, Kb, Vt);
  }
  // 5) flash attention (1024 waves)
  attn_kernel<<<(NH * (Tt / 16)) / 8, 256, 0, stream>>>(Qb, Kb, Vt, attn);
  // 6) O-proj + residual -> hres, also copy to out
  gemm_bf16_kernel<<<(Tt / 16) * (Hh / 64) / 8, 256, 0, stream>>>(
      attn, o_wt, hres, hidden, out, Tt, Hh, Hh);
  // 7) post norm
  rmsnorm_kernel<<<Tt, 256, 0, stream>>>(hres, post_norm, xn2);
  // 8) routing
  zero_counts_kernel<<<1, 32, 0, stream>>>(counts);
  router_kernel<<<Tt / 256, 256, 0, stream>>>(xn2, gate_w, gate_bias, counts,
                                              etok, ewt);
  // 9) grouped gate_up + SwiGLU
  moe_gateup_kernel<<<(Ee * (Tt / 16) * (DFF / 64)) / 8, 256, 0, stream>>>(
      xn2, gu_wt, counts, etok, actb);
  // 10) grouped down-proj, weighted scatter-add into out
  moe_down_kernel<<<(Ee * (Tt / 16) * (Hh / 64)) / 8, 256, 0, stream>>>(
      actb, dn_wt, counts, etok, ewt, out);
}
